// SDT_82489141887173
// MI455X (gfx1250) — matrix operand order repaired and checked by reading, not compiled, not run on hardware
//
#include <hip/hip_runtime.h>
#include <hip/hip_bf16.h>
#include <stdint.h>

// ---------------------------------------------------------------------------
// Soft Decision Tree forward for MI455X (gfx1250, wave32, WMMA).
//   d_in[0] = X          [16384, 1024] f32
//   d_in[1] = W_internal [1023, 1025]  f32  (col 0 = bias)
//   d_in[2] = W_leaf     [1000, 1024]  f32
//   d_out   = predictions [16384*1000] f32  ++  reg scalar (1 f32)
// ---------------------------------------------------------------------------

#define BATCH       16384
#define IN_DIM      1024
#define N_INT       1023     // internal nodes
#define N_INT_PAD   1024
#define N_LEAF      1024
#define OUT_DIM     1000
#define OUT_PAD     1024
#define DEPTH       10
#define REG_COLS    2046     // 2 + 4 + ... + 1024
#define PATH_ROWS   32       // rows handled per path-kernel block
#define PATH_BLOCKS (BATCH / PATH_ROWS)   // 512

typedef __attribute__((ext_vector_type(16))) __bf16 v16bf;
typedef __attribute__((ext_vector_type(8)))  float  v8f;

union Frag {
    v16bf v;
    uint4 q[2];
};

// ---------------------------------------------------------------------------
// Conversion kernels (f32 -> bf16, with zero padding for clean tiling)
// ---------------------------------------------------------------------------
__global__ void cvt_x_kernel(const float* __restrict__ X, __bf16* __restrict__ Xb, int n) {
    int i = blockIdx.x * blockDim.x + threadIdx.x;
    if (i < n) Xb[i] = (__bf16)X[i];
}

// W_internal [1023,1025] -> Wb [1024,1024] bf16 (bias column stripped) + bias[1024] f32
__global__ void cvt_wi_kernel(const float* __restrict__ Wi, __bf16* __restrict__ Wb,
                              float* __restrict__ bias) {
    int i = blockIdx.x * blockDim.x + threadIdx.x;   // covers 1024*1024
    int row = i >> 10, col = i & 1023;
    float v = (row < N_INT) ? Wi[(size_t)row * (IN_DIM + 1) + 1 + col] : 0.0f;
    Wb[i] = (__bf16)v;
    if (col == 0) bias[row] = (row < N_INT) ? Wi[(size_t)row * (IN_DIM + 1)] : 0.0f;
}

// W_leaf [1000,1024] -> Wb [1024,1024] bf16 (rows 1000..1023 zero)
__global__ void cvt_wl_kernel(const float* __restrict__ Wl, __bf16* __restrict__ Wb) {
    int i = blockIdx.x * blockDim.x + threadIdx.x;   // covers 1024*1024
    int row = i >> 10, col = i & 1023;
    float v = (row < OUT_DIM) ? Wl[(size_t)row * IN_DIM + col] : 0.0f;
    Wb[i] = (__bf16)v;
}

// ---------------------------------------------------------------------------
// bf16 WMMA GEMM:  C[M,N] = A[M,K] * B[N,K]^T   (K = 1024, M mult of 64, N of 32)
// One wave computes a 64x32 output tile via 4x2 v_wmma_f32_16x16x32_bf16:
//   4 A-frags + 2 B-frags (12 x b128) feed 8 WMMAs -> 1.5 b128/wmma.
// Waves of a block share the same 4 A-frag streams (consecutive tn, same tm),
// so WGP$ absorbs most A traffic.
// MODE 0: C = sigmoid(acc + bias[n])            (GEMM1 -> logits)
// MODE 1: C = acc stored if n < nvalid          (GEMM2 -> predictions)
// ---------------------------------------------------------------------------
template <int MODE>
__global__ __launch_bounds__(256) void gemm_bf16_kernel(
        const __bf16* __restrict__ A, const __bf16* __restrict__ Bm,
        const float* __restrict__ bias, float* __restrict__ C,
        int ntilesN, int ldc, int nvalid) {
    const int lane = threadIdx.x & 31;
    const int wave = threadIdx.x >> 5;
    const int w    = blockIdx.x * 8 + wave;
    const int tm   = w / ntilesN;
    const int tn   = w % ntilesN;
    const int m0   = tm * 64;
    const int n0   = tn * 32;
    const int K    = IN_DIM;

    const __bf16* Arow[4];
#pragma unroll
    for (int i = 0; i < 4; ++i)
        Arow[i] = A + (size_t)(m0 + i * 16 + (lane & 15)) * K;
    const __bf16* Brow0 = Bm + (size_t)(n0 + (lane & 15)) * K;
    const __bf16* Brow1 = Brow0 + (size_t)16 * K;
    const int ka = (lane >> 4) * 8;    // A: two 8-elem chunks at ka, ka+16
    const int kb = (lane >> 4) * 8;    // B: two 8-elem chunks at kb, kb+16

    v8f acc[4][2] = {{{}, {}}, {{}, {}}, {{}, {}}, {{}, {}}};

    for (int k0 = 0; k0 < K; k0 += 32) {
        // gfx1250 near-cache prefetch (global_prefetch_b8), 2 K-tiles ahead
        __builtin_prefetch((const void*)(Arow[0] + k0 + 64), 0, 3);
        __builtin_prefetch((const void*)(Arow[2] + k0 + 64), 0, 3);
        __builtin_prefetch((const void*)(Brow0   + k0 + 64), 0, 3);

        Frag a[4], b0, b1;
#pragma unroll
        for (int i = 0; i < 4; ++i) {
            a[i].q[0] = *(const uint4*)(Arow[i] + k0 + ka);
            a[i].q[1] = *(const uint4*)(Arow[i] + k0 + ka + 16);
        }
        b0.q[0] = *(const uint4*)(Brow0 + k0 + kb);
        b0.q[1] = *(const uint4*)(Brow0 + k0 + kb + 16);
        b1.q[0] = *(const uint4*)(Brow1 + k0 + kb);
        b1.q[1] = *(const uint4*)(Brow1 + k0 + kb + 16);

#pragma unroll
        for (int i = 0; i < 4; ++i) {
            acc[i][0] = __builtin_amdgcn_wmma_f32_16x16x32_bf16(false, a[i].v, false, b0.v,
                                                                (short)0, acc[i][0], false, false);
            acc[i][1] = __builtin_amdgcn_wmma_f32_16x16x32_bf16(false, a[i].v, false, b1.v,
                                                                (short)0, acc[i][1], false, false);
        }
    }

    // C/D layout: lane l holds column n = l%16; rows m = base + 8*(l/16) + r
    const int cn    = lane & 15;
    const int rbase = 8 * (lane >> 4);
#pragma unroll
    for (int i = 0; i < 4; ++i) {
#pragma unroll
        for (int j = 0; j < 2; ++j) {
            v8f av = acc[i][j];
            int    n  = n0 + j * 16 + cn;
            size_t mb = (size_t)(m0 + i * 16 + rbase);
            if (MODE == 0) {
                float bv = bias[n];
#pragma unroll
                for (int r = 0; r < 8; ++r) {
                    float x = av[r] + bv;
                    C[(mb + r) * (size_t)ldc + n] = 1.0f / (1.0f + __expf(-x));
                }
            } else {
                if (n < nvalid) {
#pragma unroll
                    for (int r = 0; r < 8; ++r)
                        C[(mb + r) * (size_t)ldc + n] = av[r];
                }
            }
        }
    }
}

// ---------------------------------------------------------------------------
// Tree path recursion + regularizer partials.
// Logits rows are staged into LDS with the gfx1250 async-DMA path
// (GLOBAL_LOAD_ASYNC_TO_LDS_B128, ASYNCcnt-tracked, no VGPR round-trip).
// ---------------------------------------------------------------------------
__global__ __launch_bounds__(256) void path_kernel(
        const float* __restrict__ logits,        // [BATCH, 1024] (col 1023 unused)
        __bf16* __restrict__ pathOut,            // [BATCH, 1024] bf16 leaf path probs
        float* __restrict__ partials) {          // [PATH_BLOCKS, 2*REG_COLS]
    __shared__ float sLog[N_INT_PAD];
    __shared__ float sA[N_LEAF];
    __shared__ float sB[N_LEAF];
    __shared__ float sNum[REG_COLS];
    __shared__ float sDen[REG_COLS];

    const int tid = threadIdx.x;
    for (int i = tid; i < REG_COLS; i += 256) { sNum[i] = 0.0f; sDen[i] = 0.0f; }
    __syncthreads();

    const unsigned ldsOff = (unsigned)(uintptr_t)(&sLog[tid * 4]);   // LDS byte offset

    const int row0 = blockIdx.x * PATH_ROWS;
    for (int r = 0; r < PATH_ROWS; ++r) {
        const int row = row0 + r;
        // Async global->LDS copy of one logits row (256 lanes x 16B = 4KB).
        {
            const float* gp = logits + (size_t)row * N_INT_PAD + tid * 4;
            asm volatile("global_load_async_to_lds_b128 %0, %1, off"
                         :: "v"(ldsOff), "v"((unsigned long long)(uintptr_t)gp)
                         : "memory");
            asm volatile("s_wait_asynccnt 0" ::: "memory");
        }
        __syncthreads();

        float* cur = sA;
        float* nxt = sB;
        int off = 0;
        for (int l = 0; l < DEPTH; ++l) {
            const int sz   = 2 << l;          // 2^(l+1)
            const int base = (1 << l) - 1;    // first node index of layer l
            for (int j = tid; j < sz; j += 256) {
                float s  = sLog[base + (j >> 1)];
                float lp = (j & 1) ? (1.0f - s) : s;
                float p  = (l == 0) ? 1.0f : cur[j >> 1];
                nxt[j] = p * lp;
            }
            __syncthreads();
            // num_j += lp_j * newpath[j/2] ; den_j += newpath[j/2]
            for (int j = tid; j < sz; j += 256) {
                float s  = sLog[base + (j >> 1)];
                float lp = (j & 1) ? (1.0f - s) : s;
                float pv = nxt[j >> 1];
                sNum[off + j] += lp * pv;
                sDen[off + j] += pv;
            }
            float* t = cur; cur = nxt; nxt = t;
            off += sz;
        }
        // cur holds the 1024 leaf path probabilities
        for (int j = tid; j < N_LEAF; j += 256)
            pathOut[(size_t)row * N_LEAF + j] = (__bf16)cur[j];
        __syncthreads();   // protect sLog / ping-pong buffers before next row
    }

    float* dst = partials + (size_t)blockIdx.x * (2 * REG_COLS);
    for (int i = tid; i < REG_COLS; i += 256) {
        dst[i]            = sNum[i];
        dst[REG_COLS + i] = sDen[i];
    }
}

// ---------------------------------------------------------------------------
// Fixed-order reduction of per-block partials -> regularizer scalar.
// ---------------------------------------------------------------------------
__global__ __launch_bounds__(256) void reduce_reg_kernel(
        const float* __restrict__ partials, float* __restrict__ regOut) {
    __shared__ float red[256];
    const int tid = threadIdx.x;
    float tsum = 0.0f;
    for (int col = tid; col < REG_COLS; col += 256) {
        float num = 0.0f, den = 0.0f;
        for (int b = 0; b < PATH_BLOCKS; ++b) {
            const float* p = partials + (size_t)b * (2 * REG_COLS);
            num += p[col];
            den += p[REG_COLS + col];
        }
        den += 1e-8f;
        float alpha = num / den;
        // layer of this column: off_l = 2^(l+1)-2  ->  l = floor(log2(col+2)) - 1
        int   l      = (31 - __clz(col + 2)) - 1;
        float factor = ldexpf(1.0e-3f, -l);      // LAMBDA_REG * 2^-l
        tsum += factor * (-0.5f) * (logf(alpha) + logf(1.0f - alpha));
    }
    red[tid] = tsum;
    __syncthreads();
    for (int s = 128; s > 0; s >>= 1) {
        if (tid < s) red[tid] += red[tid + s];
        __syncthreads();
    }
    if (tid == 0) regOut[0] = red[0];
}

// ---------------------------------------------------------------------------
// Host-side launcher
// ---------------------------------------------------------------------------
extern "C" void kernel_launch(void* const* d_in, const int* in_sizes, int n_in,
                              void* d_out, int out_size, void* d_ws, size_t ws_size,
                              hipStream_t stream) {
    const float* X  = (const float*)d_in[0];
    const float* Wi = (const float*)d_in[1];
    const float* Wl = (const float*)d_in[2];
    float* out = (float*)d_out;

    // Workspace layout (all offsets 256B-aligned)
    char* ws = (char*)d_ws;
    size_t off = 0;
    __bf16* Xb    = (__bf16*)(ws + off); off += (size_t)BATCH * IN_DIM * 2;        // 33.55 MB
    __bf16* Wib   = (__bf16*)(ws + off); off += (size_t)N_INT_PAD * IN_DIM * 2;    //  2.10 MB
    __bf16* Wlb   = (__bf16*)(ws + off); off += (size_t)OUT_PAD * IN_DIM * 2;      //  2.10 MB
    float*  bias  = (float*) (ws + off); off += (size_t)N_INT_PAD * 4;             //  4 KB
    float*  logit = (float*) (ws + off); off += (size_t)BATCH * N_INT_PAD * 4;     // 67.1 MB
    __bf16* pathb = (__bf16*)(ws + off); off += (size_t)BATCH * N_LEAF * 2;        // 33.55 MB
    float*  parts = (float*) (ws + off); off += (size_t)PATH_BLOCKS * 2 * REG_COLS * 4;

    // 1) Precision conversion (bf16 operands, zero-padded tiles)
    {
        int n = BATCH * IN_DIM;
        cvt_x_kernel<<<(n + 255) / 256, 256, 0, stream>>>(X, Xb, n);
        int nw = N_INT_PAD * IN_DIM;
        cvt_wi_kernel<<<(nw + 255) / 256, 256, 0, stream>>>(Wi, Wib, bias);
        cvt_wl_kernel<<<(nw + 255) / 256, 256, 0, stream>>>(Wl, Wlb);
    }

    // 2) GEMM1: logits = sigmoid(X @ Wi^T + bias)  [16384 x 1024]
    {
        int tilesM = BATCH / 64, tilesN = N_INT_PAD / 32;      // 256 x 32
        int blocks = (tilesM * tilesN) / 8;                    // 1024
        gemm_bf16_kernel<0><<<blocks, 256, 0, stream>>>(
            Xb, Wib, bias, logit, tilesN, N_INT_PAD, N_INT_PAD);
    }

    // 3) Tree path probabilities + regularizer partial sums
    path_kernel<<<PATH_BLOCKS, 256, 0, stream>>>(logit, pathb, parts);

    // 4) GEMM2: predictions = path @ Wl^T  [16384 x 1000] -> d_out
    {
        int tilesM = BATCH / 64, tilesN = OUT_PAD / 32;        // 256 x 32
        int blocks = (tilesM * tilesN) / 8;                    // 1024
        gemm_bf16_kernel<1><<<blocks, 256, 0, stream>>>(
            pathb, Wlb, nullptr, out, tilesN, OUT_DIM, OUT_DIM);
    }

    // 5) Deterministic reduction -> reg scalar at d_out[BATCH*OUT_DIM]
    reduce_reg_kernel<<<1, 256, 0, stream>>>(parts, out + (size_t)BATCH * OUT_DIM);
}
